// SlowBranch_23184233464547
// MI455X (gfx1250) — matrix operand order repaired and checked by reading, not compiled, not run on hardware
//
#include <hip/hip_runtime.h>
#include <hip/hip_bf16.h>

// ---------------------------------------------------------------------------
// GRU stack: B=64, T=16000, I=1, H=64, L=4.
// Strategy: 4 persistent workgroups (one per layer) pipelined over L2 flags.
// Each step: hg/xg GEMMs via v_wmma_f32_16x16x32_f16, gates in registers,
// h state re-layout through a 64x72 f16 LDS tile (ds_load_b128 A-fragments).
// ---------------------------------------------------------------------------

#define TSEQ 16000
#define NB   64
#define NH   64
#define NL   4
#define SEQ_ELEMS ((size_t)TSEQ * NB * NH)   // 65,536,000 halves per stream buffer
#define SH_LD 72                              // padded LDS leading dim (16B-aligned rows)

typedef __attribute__((ext_vector_type(16))) _Float16 v16h;
typedef __attribute__((ext_vector_type(8)))  _Float16 v8h;
typedef __attribute__((ext_vector_type(8)))  float    v8f;

__device__ __forceinline__ v8f wmma_f16(v16h a, v16h b, v8f c) {
  // D = A(16x32 f16) * B(32x16 f16) + C(16x16 f32)
  return __builtin_amdgcn_wmma_f32_16x16x32_f16(false, a, false, b, (short)0, c,
                                                false, false);
}

__device__ __forceinline__ float sigf(float x) {
  x = fminf(fmaxf(x, -30.f), 30.f);
  return 1.f / (1.f + __expf(-x));
}
__device__ __forceinline__ float tanhfast(float x) {
  x = fminf(fmaxf(x, -15.f), 15.f);
  float e = __expf(-2.f * x);
  return (1.f - e) / (1.f + e);
}

// B-fragment (32x16, f16) for lane `lane`, gathered from row-major f32 weight
// W[3H][H] used transposed: B[k][n] = W[n0+n][k0+k].
// Assumed layout: lanes 0-15 hold col n=lane, K = k0+{0..7,16..23} ; lanes 16-31 hold
// col n=lane-16, K = k0+{8..15,24..31}.
__device__ __forceinline__ v16h load_bfrag(const float* __restrict__ W,
                                           int n0, int k0, int lane) {
  const int n  = n0 + (lane & 15);
  const int kb = k0 + ((lane >> 4) << 3);
  const float* p = W + n * NH + kb;
  v16h r;
#pragma unroll
  for (int i = 0; i < 16; ++i) r[i] = (_Float16)p[(i & 7) + ((i >> 3) << 4)];
  return r;
}

// A-fragment (16x32, f16) from a row-major f16 tile with leading dim `ld`.
// ISA layout: lanes 0-15: row=lane, K = {0..7,16..23}; lanes 16-31: row=lane-16,
// K = {8..15,24..31}.  kb = k0 + (lane>=16 ? 8 : 0).
__device__ __forceinline__ v16h load_afrag(const _Float16* __restrict__ base,
                                           int ld, int row, int kb) {
  const v8h lo = *(const v8h*)(base + (size_t)row * ld + kb);
  const v8h hi = *(const v8h*)(base + (size_t)row * ld + kb + 16);
  return __builtin_shufflevector(lo, hi, 0, 1, 2, 3, 4, 5, 6, 7,
                                 8, 9, 10, 11, 12, 13, 14, 15);
}

// ---------------------------------------------------------------------------
// Kernel 1: input projection h0[t][b][h] = relu(x[b][t] * W_in[h] + b_in[h]),
// stored f16 in stream layout [t][b][h]. Also resets pipeline flags.
// ---------------------------------------------------------------------------
__global__ void proj_relu_kernel(const float* __restrict__ x,
                                 const float* __restrict__ W_in,
                                 const float* __restrict__ b_in,
                                 _Float16* __restrict__ bufA,
                                 int* __restrict__ prog) {
  if (blockIdx.x == 0 && threadIdx.x < NL)
    prog[threadIdx.x] = 0;
  size_t idx = (size_t)blockIdx.x * blockDim.x + threadIdx.x;
  if (idx >= SEQ_ELEMS) return;
  const int h = (int)(idx & 63);
  const int b = (int)((idx >> 6) & 63);
  const int t = (int)(idx >> 12);
  float v = x[(size_t)b * TSEQ + t] * W_in[h] + b_in[h];
  bufA[idx] = (_Float16)fmaxf(v, 0.f);
}

// ---------------------------------------------------------------------------
// Kernel 2: persistent pipelined GRU scan. Block l == layer l.
// ---------------------------------------------------------------------------
__global__ void __launch_bounds__(512, 1)
gru_pipeline(_Float16* __restrict__ bufA, _Float16* __restrict__ bufB,
             const float* __restrict__ W_ih, const float* __restrict__ W_hh,
             const float* __restrict__ b_ih, const float* __restrict__ b_hh,
             const float* __restrict__ W_out, const float* __restrict__ b_out,
             float* __restrict__ out, int* __restrict__ prog) {
  __shared__ _Float16 sH[NB * SH_LD];   // current hidden state, f16, padded

  const int l = blockIdx.x;
  const _Float16* inSeq = (l & 1) ? bufB : bufA;
  _Float16* outSeq      = (l & 1) ? bufA : bufB;
  const float* Wih = W_ih + (size_t)l * 3 * NH * NH;
  const float* Whh = W_hh + (size_t)l * 3 * NH * NH;

  const int lane = threadIdx.x & 31;
  const int wave = threadIdx.x >> 5;     // 0..15
  const int m0   = (wave & 3) * 16;      // batch-row tile
  const int g    = wave >> 2;            // gate-column group 0..3
  const int jc   = g * 16 + (lane & 15); // hidden column owned for gates
  const int ksel = lane >> 4;            // 0: lanes 0-15, 1: lanes 16-31

  // --- per-layer loop-invariant state: biases + 12 B-fragments in VGPRs ----
  const float xbr = b_ih[(size_t)l * 3 * NH + jc];
  const float xbz = b_ih[(size_t)l * 3 * NH + NH + jc];
  const float xbn = b_ih[(size_t)l * 3 * NH + 2 * NH + jc];
  const float hbr = b_hh[(size_t)l * 3 * NH + jc];
  const float hbz = b_hh[(size_t)l * 3 * NH + NH + jc];
  const float hbn = b_hh[(size_t)l * 3 * NH + 2 * NH + jc];

  const int nR = g * 16, nZ = NH + g * 16, nN = 2 * NH + g * 16;
  const v16h biR0 = load_bfrag(Wih, nR, 0, lane), biR1 = load_bfrag(Wih, nR, 32, lane);
  const v16h biZ0 = load_bfrag(Wih, nZ, 0, lane), biZ1 = load_bfrag(Wih, nZ, 32, lane);
  const v16h biN0 = load_bfrag(Wih, nN, 0, lane), biN1 = load_bfrag(Wih, nN, 32, lane);
  const v16h bhR0 = load_bfrag(Whh, nR, 0, lane), bhR1 = load_bfrag(Whh, nR, 32, lane);
  const v16h bhZ0 = load_bfrag(Whh, nZ, 0, lane), bhZ1 = load_bfrag(Whh, nZ, 32, lane);
  const v16h bhN0 = load_bfrag(Whh, nN, 0, lane), bhN1 = load_bfrag(Whh, nN, 32, lane);

  // h(t=0) = 0
  for (int i = threadIdx.x; i < NB * SH_LD; i += 512) sH[i] = (_Float16)0.f;
  float hold[8];
#pragma unroll
  for (int i = 0; i < 8; ++i) hold[i] = 0.f;
  __syncthreads();

  const int kb = ksel * 8;          // A-fragment low-group offset
  const int arow = m0 + (lane & 15);

  for (int t = 0; t < TSEQ; ++t) {
    // wait for producer layer (skipped for layer 0)
    if (l > 0) {
      if (threadIdx.x == 0) {
        while (__hip_atomic_load(&prog[l - 1], __ATOMIC_ACQUIRE,
                                 __HIP_MEMORY_SCOPE_AGENT) <= t)
          __builtin_amdgcn_s_sleep(2);
      }
      __syncthreads();
    }

    // layer-input A-fragments (global, f16 [t][b][h]) + prefetch ahead
    const _Float16* xt = inSeq + (size_t)t * (NB * NH);
    const int tp = (t + 8 < TSEQ) ? t + 8 : TSEQ - 1;
    __builtin_prefetch(inSeq + (size_t)tp * (NB * NH) + (size_t)arow * NH, 0, 0);
    const v16h xa0 = load_afrag(xt, NH, arow, kb);
    const v16h xa1 = load_afrag(xt, NH, arow, 32 + kb);

    // hidden-state A-fragments from LDS (prev step's stores synced above / end)
    const v16h ha0 = load_afrag(sH, SH_LD, arow, kb);
    const v16h ha1 = load_afrag(sH, SH_LD, arow, 32 + kb);
    __syncthreads();   // all reads of sH complete before anyone overwrites it

    // GEMMs: r/z accumulate x-part and h-part together; n kept split.
    v8f zero = {};
    v8f accR = wmma_f16(ha1, bhR1, wmma_f16(ha0, bhR0,
               wmma_f16(xa1, biR1, wmma_f16(xa0, biR0, zero))));
    v8f accZ = wmma_f16(ha1, bhZ1, wmma_f16(ha0, bhZ0,
               wmma_f16(xa1, biZ1, wmma_f16(xa0, biZ0, zero))));
    v8f accXN = wmma_f16(xa1, biN1, wmma_f16(xa0, biN0, zero));
    v8f accHN = wmma_f16(ha1, bhN1, wmma_f16(ha0, bhN0, zero));

    // gates, fully register-local; update h and publish the step
#pragma unroll
    for (int i = 0; i < 8; ++i) {
      const float r = sigf(accR[i] + xbr + hbr);
      const float z = sigf(accZ[i] + xbz + hbz);
      const float n = tanhfast(accXN[i] + xbn + r * (accHN[i] + hbn));
      const float hn = (1.f - z) * n + z * hold[i];
      hold[i] = hn;
      const _Float16 hh = (_Float16)hn;
      const int row = m0 + i + (ksel << 3);       // D-layout row mapping
      sH[row * SH_LD + jc] = hh;
      outSeq[(size_t)t * (NB * NH) + (size_t)row * NH + jc] = hh;
    }

    __threadfence();     // own stores device-visible before flag release
    __syncthreads();
    if (threadIdx.x == 0)
      __hip_atomic_store(&prog[l], t + 1, __ATOMIC_RELEASE,
                         __HIP_MEMORY_SCOPE_AGENT);
  }

  // final FC on last layer's block: out[b][j] = h_T[b] . W_out[j] + b_out[j]
  if (l == NL - 1) {
    __syncthreads();
    for (int e = threadIdx.x; e < NB * NH; e += 512) {
      const int b = e >> 6, j = e & 63;
      float acc = b_out[j];
#pragma unroll 8
      for (int k = 0; k < NH; ++k)
        acc += (float)sH[b * SH_LD + k] * W_out[j * NH + k];
      out[e] = acc;
    }
  }
}

extern "C" void kernel_launch(void* const* d_in, const int* in_sizes, int n_in,
                              void* d_out, int out_size, void* d_ws, size_t ws_size,
                              hipStream_t stream) {
  const float* x     = (const float*)d_in[0];
  const float* W_in  = (const float*)d_in[1];
  const float* b_in  = (const float*)d_in[2];
  const float* W_ih  = (const float*)d_in[3];
  const float* W_hh  = (const float*)d_in[4];
  const float* b_ih  = (const float*)d_in[5];
  const float* b_hh  = (const float*)d_in[6];
  const float* W_out = (const float*)d_in[7];
  const float* b_out = (const float*)d_in[8];

  _Float16* bufA = (_Float16*)d_ws;
  _Float16* bufB = bufA + SEQ_ELEMS;
  int* prog = (int*)(bufB + SEQ_ELEMS);

  const int threads = 256;
  const int blocks = (int)((SEQ_ELEMS + threads - 1) / threads);
  proj_relu_kernel<<<blocks, threads, 0, stream>>>(x, W_in, b_in, bufA, prog);

  gru_pipeline<<<NL, 512, 0, stream>>>(bufA, bufB, W_ih, W_hh, b_ih, b_hh,
                                       W_out, b_out, (float*)d_out, prog);
}
